// MoELayer_76974403879710
// MI455X (gfx1250) — compile-verified, matrix operand order checked by reading, not run on hardware
//
#include <hip/hip_runtime.h>
#include <hip/hip_bf16.h>
#include <math.h>

// ---------------- problem constants (match reference) ----------------
constexpr int NTOK = 4 * 2048;   // B*T = 8192
constexpr int DD   = 1024;       // model dim
constexpr int HH   = 2048;       // hidden dim
constexpr int EE   = 8;          // experts
constexpr int MT   = 32;         // token tile per expert workgroup
constexpr float LN_EPS = 1e-5f;

typedef __bf16 bf16_t;
typedef __attribute__((ext_vector_type(16))) __bf16 v16bf;
typedef __attribute__((ext_vector_type(8)))  __bf16 v8bf;
typedef __attribute__((ext_vector_type(8)))  float  v8f;

// Build a 16-element bf16 WMMA fragment from two contiguous 8-element (16B)
// loads at p and p+16. With p = row_base + k + 8*(lane>=16), this matches the
// CDNA5 16-bit A/B VGPR layout (K = k..k+7 in v0-3, K = k+16..k+23 in v4-7,
// upper-half lanes shifted by 8).
__device__ __forceinline__ v16bf load_frag16(const bf16_t* p) {
    v8bf lo = *(const v8bf*)(p);
    v8bf hi = *(const v8bf*)(p + 16);
    return __builtin_shufflevector(lo, hi, 0,1,2,3,4,5,6,7,8,9,10,11,12,13,14,15);
}

__device__ __forceinline__ float gelu_exact(float v) {
    return 0.5f * v * (1.0f + erff(v * 0.70710678118654752f));
}

#define WMMA_BF16(A, B, C) \
    __builtin_amdgcn_wmma_f32_16x16x32_bf16(false, (A), false, (B), (short)0, (C), false, false)

// ---------------- pass 1: weight/activation convert to bf16 ----------------
// W1 [E][D][H] f32 -> W1T [E][H][D] bf16   (B-fragments contiguous along K=D)
__global__ void convert_w1_kernel(const float* __restrict__ W1, bf16_t* __restrict__ W1T) {
    size_t idx = (size_t)blockIdx.x * blockDim.x + threadIdx.x;  // over E*H*D
    int d = (int)(idx % DD);
    size_t t = idx / DD;
    int h = (int)(t % HH);
    int e = (int)(t / HH);
    W1T[idx] = (bf16_t)W1[((size_t)e * DD + d) * HH + h];
}
// W2 [E][H][D] f32 -> W2T [E][D][H] bf16   (B-fragments contiguous along K=H)
__global__ void convert_w2_kernel(const float* __restrict__ W2, bf16_t* __restrict__ W2T) {
    size_t idx = (size_t)blockIdx.x * blockDim.x + threadIdx.x;  // over E*D*H
    int h = (int)(idx % HH);
    size_t t = idx / HH;
    int d = (int)(t % DD);
    int e = (int)(t / DD);
    W2T[idx] = (bf16_t)W2[((size_t)e * HH + h) * DD + d];
}
// x f32 -> xbf bf16 (enables direct global->LDS async staging in expert pass)
__global__ void convert_x_kernel(const float* __restrict__ x, bf16_t* __restrict__ xbf) {
    size_t idx = (size_t)blockIdx.x * blockDim.x + threadIdx.x;  // over N*D
    xbf[idx] = (bf16_t)x[idx];
}

// ---------------- pass 2: router (top-2 gate + dispatch + aux stats) --------
__global__ void router_kernel(const float* __restrict__ x,
                              const float* __restrict__ gW,
                              const float* __restrict__ gb,
                              int*   __restrict__ counts,
                              int*   __restrict__ btok,
                              float* __restrict__ bgate,
                              float* __restrict__ imp_sum,
                              float* __restrict__ ent_sum) {
    int n = blockIdx.x * blockDim.x + threadIdx.x;
    if (n >= NTOK) return;
    float l[EE];
#pragma unroll
    for (int e = 0; e < EE; ++e) l[e] = gb[e];
    const float* xr = x + (size_t)n * DD;
    for (int d = 0; d < DD; ++d) {
        float xv = xr[d];
        const float* gr = gW + (size_t)d * EE;   // contiguous 8 floats
#pragma unroll
        for (int e = 0; e < EE; ++e) l[e] = fmaf(xv, gr[e], l[e]);
    }
    // top-2
    int bi = 0; float bv = l[0];
#pragma unroll
    for (int e = 1; e < EE; ++e) if (l[e] > bv) { bv = l[e]; bi = e; }
    int si = (bi == 0) ? 1 : 0; float sv = l[si];
#pragma unroll
    for (int e = 0; e < EE; ++e) if (e != bi && l[e] > sv) { sv = l[e]; si = e; }
    // softmax over the top-2 values
    float e2 = __expf(sv - bv);
    float g1 = 1.0f / (1.0f + e2);
    float g2 = e2 * g1;
    // dispatch to expert buckets
    int p1 = atomicAdd(&counts[bi], 1);
    btok[bi * NTOK + p1] = n; bgate[bi * NTOK + p1] = g1;
    int p2 = atomicAdd(&counts[si], 1);
    btok[si * NTOK + p2] = n; bgate[si * NTOK + p2] = g2;
    // full softmax for importance / entropy
    float p[EE], s = 0.0f;
#pragma unroll
    for (int e = 0; e < EE; ++e) { p[e] = __expf(l[e] - bv); s += p[e]; }
    float inv = 1.0f / s, ent = 0.0f;
#pragma unroll
    for (int e = 0; e < EE; ++e) {
        p[e] *= inv;
        ent -= p[e] * __logf(p[e] + 1e-8f);
        atomicAdd(&imp_sum[e], p[e]);
    }
    atomicAdd(ent_sum, ent);
}

// ---------------- pass 3: fused expert MLP (WMMA bf16) ----------------------
// block = 256 threads (8 waves), one 32-token tile of one expert.
// LDS: X tile (32x1024 bf16) + GeLU intermediate (32x2048 bf16) = 192KB
// (CDNA5 WGP has 320KB). X is staged with global_load_async_to_lds_b128.
// Register blocking: 2 row-blocks x 4 column tiles -> 8 WMMA per A-frag pair.
__global__ void expert_mlp_kernel(const bf16_t* __restrict__ xbf,
                                  const bf16_t* __restrict__ W1T,
                                  const float*  __restrict__ b1,
                                  const bf16_t* __restrict__ W2T,
                                  const float*  __restrict__ b2,
                                  const int*    __restrict__ counts,
                                  const int*    __restrict__ btok,
                                  const float*  __restrict__ bgate,
                                  float*        __restrict__ moe_out) {
    const int e  = blockIdx.y;
    const int cnt = counts[e];
    const int t0 = blockIdx.x * MT;
    if (t0 >= cnt) return;                    // uniform per-block exit

    extern __shared__ char smem_raw[];
    bf16_t* sX   = (bf16_t*)smem_raw;         // [MT][DD]  (LDS byte offset 0)
    bf16_t* sH   = sX + MT * DD;              // [MT][HH]
    int*    sTok = (int*)(sH + MT * HH);      // [MT]
    float*  sGat = (float*)(sTok + MT);       // [MT]

    const int tid = threadIdx.x;
    if (tid < MT) {
        int i = t0 + tid;
        if (i < cnt) { sTok[tid] = btok[e * NTOK + i]; sGat[tid] = bgate[e * NTOK + i]; }
        else         { sTok[tid] = 0;                  sGat[tid] = 0.0f; }
    }
    __syncthreads();

    // stage gathered X rows straight into LDS: async global->LDS 16B copies.
    // Dynamic LDS segment starts at offset 0 (no static LDS in this kernel),
    // so the LDS destination operand is just the byte offset into sX.
    {
        constexpr int CHUNKS = MT * DD / 8;            // 8 bf16 = 16B per chunk
        for (int i = tid; i < CHUNKS; i += 256) {
            int r = i / (DD / 8);
            int c = (i % (DD / 8)) * 8;
            unsigned lds_off = (unsigned)(((size_t)r * DD + c) * sizeof(bf16_t));
            unsigned long long gaddr =
                (unsigned long long)(const void*)(xbf + (size_t)sTok[r] * DD + c);
            asm volatile("global_load_async_to_lds_b128 %0, %1, off"
                         :: "v"(lds_off), "v"(gaddr) : "memory");
        }
        asm volatile("s_wait_asynccnt 0" ::: "memory");
    }
    __syncthreads();

    const int lane = tid & 31;
    const int wave = tid >> 5;
    const int m    = lane & 15;
    const int half = lane >> 4;
    const int koff = 8 * half;                // lane-half K shift per ISA layout
    const int mrow0 = m, mrow1 = 16 + m;

    // ---- GEMM1: H = gelu(X @ W1 + b1); wave covers 256 of 2048 columns ----
    const bf16_t* w1  = W1T + (size_t)e * HH * DD;  // [HH][DD]
    const float*  b1e = b1 + (size_t)e * HH;
    for (int cb = 0; cb < 4; ++cb) {                // 4 blocks of 4 col tiles
        const int nb = wave * (HH / 8) + cb * 64;
        __builtin_prefetch(w1 + (size_t)(nb + m) * DD, 0, 1);
        v8f acc[8] = {};
        for (int k = 0; k < DD; k += 32) {
            const int ko = k + koff;
            v16bf a0 = load_frag16(sX + (size_t)mrow0 * DD + ko);
            v16bf a1 = load_frag16(sX + (size_t)mrow1 * DD + ko);
#pragma unroll
            for (int t = 0; t < 4; ++t) {
                v16bf bb = load_frag16(w1 + (size_t)(nb + 16 * t + m) * DD + ko);
                acc[2 * t]     = WMMA_BF16(a0, bb, acc[2 * t]);
                acc[2 * t + 1] = WMMA_BF16(a1, bb, acc[2 * t + 1]);
            }
        }
#pragma unroll
        for (int t = 0; t < 4; ++t) {
            const int col  = nb + 16 * t + m;
            const float bias = b1e[col];
#pragma unroll
            for (int j = 0; j < 8; ++j) {
                int row = j + 8 * half;
                sH[(size_t)row        * HH + col] = (bf16_t)gelu_exact(acc[2 * t][j]     + bias);
                sH[(size_t)(row + 16) * HH + col] = (bf16_t)gelu_exact(acc[2 * t + 1][j] + bias);
            }
        }
    }
    __syncthreads();

    // ---- GEMM2: Y = H @ W2 + b2, gate-weighted scatter-add ----
    const bf16_t* w2  = W2T + (size_t)e * DD * HH;  // [DD][HH]
    const float*  b2e = b2 + (size_t)e * DD;
    for (int cb = 0; cb < 2; ++cb) {                // 2 blocks of 4 col tiles
        const int nb = wave * (DD / 8) + cb * 64;
        __builtin_prefetch(w2 + (size_t)(nb + m) * HH, 0, 1);
        v8f acc[8] = {};
        for (int k = 0; k < HH; k += 32) {
            const int ko = k + koff;
            v16bf a0 = load_frag16(sH + (size_t)mrow0 * HH + ko);
            v16bf a1 = load_frag16(sH + (size_t)mrow1 * HH + ko);
#pragma unroll
            for (int t = 0; t < 4; ++t) {
                v16bf bb = load_frag16(w2 + (size_t)(nb + 16 * t + m) * HH + ko);
                acc[2 * t]     = WMMA_BF16(a0, bb, acc[2 * t]);
                acc[2 * t + 1] = WMMA_BF16(a1, bb, acc[2 * t + 1]);
            }
        }
#pragma unroll
        for (int t = 0; t < 4; ++t) {
            const int col  = nb + 16 * t + m;
            const float bias = b2e[col];
#pragma unroll
            for (int j = 0; j < 8; ++j) {
                int r0 = j + 8 * half;
                int r1 = r0 + 16;
                float g0 = sGat[r0], g1 = sGat[r1];
                if (g0 != 0.0f)
                    atomicAdd(&moe_out[(size_t)sTok[r0] * DD + col], (acc[2 * t][j]     + bias) * g0);
                if (g1 != 0.0f)
                    atomicAdd(&moe_out[(size_t)sTok[r1] * DD + col], (acc[2 * t + 1][j] + bias) * g1);
            }
        }
    }
}

// ---------------- pass 4: residual + LayerNorm ------------------------------
__global__ void ln_kernel(const float* __restrict__ x,
                          const float* __restrict__ moe,
                          const float* __restrict__ gamma,
                          const float* __restrict__ beta,
                          float* __restrict__ out) {
    const int n   = blockIdx.x;
    const int tid = threadIdx.x;
    __shared__ float rs[256], rq[256];
    const float* xr = x   + (size_t)n * DD;
    const float* mr = moe + (size_t)n * DD;
    float z[DD / 256];
    float s = 0.0f, q = 0.0f;
#pragma unroll
    for (int i = 0; i < DD / 256; ++i) {
        int d = tid + i * 256;
        float v = xr[d] + mr[d];
        z[i] = v; s += v; q += v * v;
    }
    rs[tid] = s; rq[tid] = q;
    __syncthreads();
    for (int off = 128; off > 0; off >>= 1) {
        if (tid < off) { rs[tid] += rs[tid + off]; rq[tid] += rq[tid + off]; }
        __syncthreads();
    }
    const float mu  = rs[0] * (1.0f / DD);
    const float var = rq[0] * (1.0f / DD) - mu * mu;
    const float inv = rsqrtf(var + LN_EPS);
#pragma unroll
    for (int i = 0; i < DD / 256; ++i) {
        int d = tid + i * 256;
        out[(size_t)n * DD + d] = (z[i] - mu) * inv * gamma[d] + beta[d];
    }
}

// ---------------- pass 5: aux losses ----------------------------------------
__global__ void finalize_kernel(const int* __restrict__ counts,
                                const float* __restrict__ imp_sum,
                                const float* __restrict__ ent_sum,
                                float* __restrict__ tail) {
    if (threadIdx.x != 0 || blockIdx.x != 0) return;
    const float invN = 1.0f / (float)NTOK;
    float bal = 0.0f, util = 0.0f, loadv[EE], imp[EE];
#pragma unroll
    for (int e = 0; e < EE; ++e) {
        loadv[e] = (float)counts[e] * invN;
        imp[e]   = imp_sum[e] * invN;
        bal  += imp[e] * loadv[e];
        util -= loadv[e] * __logf(loadv[e] + 1e-8f);
    }
    tail[0] = (float)EE * bal;        // balance_loss
    tail[1] = ent_sum[0] * invN;      // entropy
    tail[2] = util;                   // utilization_entropy
#pragma unroll
    for (int e = 0; e < EE; ++e) tail[3 + e]  = loadv[e];
#pragma unroll
    for (int e = 0; e < EE; ++e) tail[11 + e] = imp[e];
}

// ---------------- host orchestration ----------------------------------------
extern "C" void kernel_launch(void* const* d_in, const int* in_sizes, int n_in,
                              void* d_out, int out_size, void* d_ws, size_t ws_size,
                              hipStream_t stream) {
    const float* x      = (const float*)d_in[0];
    const float* gate_W = (const float*)d_in[1];
    const float* gate_b = (const float*)d_in[2];
    const float* W1     = (const float*)d_in[3];
    const float* b1     = (const float*)d_in[4];
    const float* W2     = (const float*)d_in[5];
    const float* b2     = (const float*)d_in[6];
    const float* ln_g   = (const float*)d_in[7];
    const float* ln_b   = (const float*)d_in[8];
    float* out = (float*)d_out;

    // workspace layout
    char* ws = (char*)d_ws;
    const size_t szWT  = (size_t)EE * HH * DD * sizeof(bf16_t);   // 32 MiB each
    const size_t szXbf = (size_t)NTOK * DD * sizeof(bf16_t);      // 16 MiB
    const size_t szMoe = (size_t)NTOK * DD * sizeof(float);       // 32 MiB
    const size_t szB   = (size_t)EE * NTOK * sizeof(int);         // 256 KiB
    bf16_t* W1T     = (bf16_t*)(ws);
    bf16_t* W2T     = (bf16_t*)(ws + szWT);
    bf16_t* xbf     = (bf16_t*)(ws + 2 * szWT);
    float*  moe_out = (float*) (ws + 2 * szWT + szXbf);
    int*    btok    = (int*)   (ws + 2 * szWT + szXbf + szMoe);
    float*  bgate   = (float*) ((char*)btok + szB);
    int*    counts  = (int*)   ((char*)bgate + szB);
    float*  imp_sum = (float*) ((char*)counts + EE * sizeof(int));
    float*  ent_sum = (float*) ((char*)imp_sum + EE * sizeof(float));

    // zero accumulators (graph-capture-safe async memsets)
    hipMemsetAsync(moe_out, 0, szMoe, stream);
    hipMemsetAsync(counts, 0, EE * sizeof(int) + (EE + 1) * sizeof(float), stream);

    // pass 1: bf16 transposed weights + bf16 activations
    {
        size_t n1 = (size_t)EE * HH * DD;
        convert_w1_kernel<<<dim3((unsigned)(n1 / 256)), dim3(256), 0, stream>>>(W1, W1T);
        convert_w2_kernel<<<dim3((unsigned)(n1 / 256)), dim3(256), 0, stream>>>(W2, W2T);
        convert_x_kernel<<<dim3((unsigned)((size_t)NTOK * DD / 256)), dim3(256), 0, stream>>>(x, xbf);
    }
    // pass 2: router + dispatch
    router_kernel<<<dim3(NTOK / 256), dim3(256), 0, stream>>>(
        x, gate_W, gate_b, counts, btok, bgate, imp_sum, ent_sum);
    // pass 3: fused expert MLP (WMMA); 192KB+ dynamic LDS
    {
        const size_t lds_bytes = (size_t)(MT * DD + MT * HH) * sizeof(bf16_t)
                               + MT * sizeof(int) + MT * sizeof(float);
        dim3 grid(NTOK / MT, EE);
        expert_mlp_kernel<<<grid, dim3(256), lds_bytes, stream>>>(
            xbf, W1T, b1, W2T, b2, counts, btok, bgate, moe_out);
    }
    // pass 4: residual + layernorm into d_out
    ln_kernel<<<dim3(NTOK), dim3(256), 0, stream>>>(x, moe_out, ln_g, ln_b, out);
    // pass 5: aux losses into d_out tail
    finalize_kernel<<<dim3(1), dim3(32), 0, stream>>>(
        counts, imp_sum, ent_sum, out + (size_t)NTOK * DD);
}
